// MultiHeadAttention_23321672417657
// MI455X (gfx1250) — matrix operand order repaired and checked by reading, not compiled, not run on hardware
//
#include <hip/hip_runtime.h>

#define S_LEN 2048
#define DM    1024
#define NH    16
#define DK    64

typedef __attribute__((ext_vector_type(16))) __bf16        v16bf;
typedef __attribute__((ext_vector_type(8)))  float         v8f;
typedef __attribute__((ext_vector_type(4)))  unsigned int  u32x4;

struct F2 { u32x4 lo, hi; };

static __device__ __forceinline__ unsigned short f2bf(float f) {
    unsigned int u = __float_as_uint(f);
    u += 0x7FFFu + ((u >> 16) & 1u);          // round-to-nearest-even
    return (unsigned short)(u >> 16);
}

static __device__ __forceinline__ v16bf ld_frag(const unsigned short* p0,
                                                const unsigned short* p1) {
    F2 t;
    t.lo = *(const u32x4*)p0;
    t.hi = *(const u32x4*)p1;
    return __builtin_bit_cast(v16bf, t);
}

static __device__ __forceinline__ v8f vzero() {
    v8f z;
#pragma unroll
    for (int i = 0; i < 8; ++i) z[i] = 0.f;
    return z;
}

// CDNA5 async global->LDS copy (ASYNCcnt-tracked, 16B per lane).
// ldsoff: wave-relative LDS byte address (low 32 bits of the flat shared ptr).
static __device__ __forceinline__ void async_cp_b128(const void* lds_dst,
                                                     const void* gsrc) {
    unsigned loff = (unsigned)(unsigned long long)lds_dst;
    asm volatile("global_load_async_to_lds_b128 %0, %1, off"
                 :: "v"(loff), "v"(gsrc) : "memory");
}
static __device__ __forceinline__ void wait_async0() {
    asm volatile("s_wait_asynccnt 0x0" ::: "memory");
}

// ---------------------------------------------------------------------------
// GEMM: Y[M=4096][N=1024] = X[4096][1024] @ W[1024][1024] + bias
// MODE 0: X fp32, out bf16 per-head [B*H][S][64]             (Q and K)
// MODE 2: X fp32, out bf16 per-head transposed [B*H][64][S]  (V)
// MODE 3: X bf16, out fp32 row-major [4096][1024]            (final proj)
// Block: 128 threads (4 waves), tile 128x128, K-chunks of 32.
// ---------------------------------------------------------------------------
#define LDA 40   // padded row length (ushorts), 16B aligned rows

template<int MODE>
__global__ __launch_bounds__(128)
void gemm128(const void* __restrict__ Xv, const float* __restrict__ W,
             const float* __restrict__ bias, void* __restrict__ Outv) {
    __shared__ unsigned short lA[128][LDA];
    __shared__ unsigned short lB[128][LDA];   // stored transposed: lB[n][k]

    const int t    = threadIdx.x;
    const int lane = t & 31, w = t >> 5;
    const int half = lane >> 4, lm = lane & 15;
    const int m0   = blockIdx.y * 128;
    const int n0   = blockIdx.x * 128;
    const int m0w  = (w >> 1) * 64;
    const int n0w  = (w & 1) * 64;

    v8f acc[4][4];
#pragma unroll
    for (int r = 0; r < 4; ++r)
#pragma unroll
        for (int c = 0; c < 4; ++c) acc[r][c] = vzero();

    for (int kt = 0; kt < DM / 32; ++kt) {
        const int k0 = kt * 32;
        __syncthreads();
        // --- load A tile 128x32 (row-major) ---
        if (MODE == 3) {
            const unsigned short* X = (const unsigned short*)Xv;
#pragma unroll
            for (int i = 0; i < 32; ++i) {
                int e = t + i * 128, m = e >> 5, k = e & 31;
                lA[m][k] = X[(size_t)(m0 + m) * DM + k0 + k];
            }
        } else {
            const float* X = (const float*)Xv;
#pragma unroll
            for (int i = 0; i < 32; ++i) {
                int e = t + i * 128, m = e >> 5, k = e & 31;
                lA[m][k] = f2bf(X[(size_t)(m0 + m) * DM + k0 + k]);
            }
        }
        // --- load W tile 32x128, store transposed lB[n][k] ---
#pragma unroll
        for (int i = 0; i < 32; ++i) {
            int e = t + i * 128, k = e >> 7, n = e & 127;
            lB[n][k] = f2bf(W[(size_t)(k0 + k) * DM + n0 + n]);
        }
        __syncthreads();

        // prefetch next K-chunk into cache while WMMAs run
        if (kt + 1 < DM / 32) {
            const int kn = k0 + 32;
            if (MODE == 3)
                __builtin_prefetch(((const unsigned short*)Xv) +
                                   (size_t)(m0 + t) * DM + kn, 0, 1);
            else
                __builtin_prefetch(((const float*)Xv) +
                                   (size_t)(m0 + t) * DM + kn, 0, 1);
            __builtin_prefetch(W + (size_t)(kn + (t & 31)) * DM + n0 +
                               ((t >> 5) << 5), 0, 1);
        }

        v16bf af[4], bf[4];
#pragma unroll
        for (int r = 0; r < 4; ++r) {
            const unsigned short* row = &lA[m0w + 16 * r + lm][0];
            af[r] = ld_frag(row + half * 8, row + 16 + half * 8);
        }
#pragma unroll
        for (int c = 0; c < 4; ++c) {
            const unsigned short* row = &lB[n0w + 16 * c + lm][0];
            bf[c] = ld_frag(row + half * 8, row + 16 + half * 8);
        }
#pragma unroll
        for (int r = 0; r < 4; ++r)
#pragma unroll
            for (int c = 0; c < 4; ++c)
                acc[r][c] = __builtin_amdgcn_wmma_f32_16x16x32_bf16(
                    false, af[r], false, bf[c], (short)0, acc[r][c], false, false);
    }

    // --- epilogue ---
#pragma unroll
    for (int r = 0; r < 4; ++r)
#pragma unroll
        for (int c = 0; c < 4; ++c)
#pragma unroll
            for (int j = 0; j < 8; ++j) {
                int mg = m0 + m0w + 16 * r + j + 8 * half;
                int ng = n0 + n0w + 16 * c + lm;
                float v = acc[r][c][j] + bias[ng];
                if (MODE == 3) {
                    ((float*)Outv)[(size_t)mg * DM + ng] = v;
                } else {
                    int b = mg >> 11, s = mg & (S_LEN - 1);
                    int h = ng >> 6,  d = ng & 63;
                    unsigned short* O = (unsigned short*)Outv;
                    if (MODE == 2)
                        O[((size_t)((b * NH + h) * DK + d)) * S_LEN + s] = f2bf(v);
                    else
                        O[((size_t)((b * NH + h) * S_LEN + s)) * DK + d] = f2bf(v);
                }
            }
}

// ---------------------------------------------------------------------------
// Flash attention: grid (S/64, B*H), 128 threads (4 waves).
// Double-buffered K/V tiles staged with async global->LDS copies (ASYNCcnt),
// prefetching tile skt+1 while tile skt runs through the WMMAs.
// Q,K: [bh][S][64] bf16.  VT: [bh][64][S] bf16.  O: [B*S][1024] bf16.
// ---------------------------------------------------------------------------
#define LDF 72   // padded row (ushorts), rows 144B -> 16B aligned

__global__ __launch_bounds__(128)
void flash_attn(const unsigned short* __restrict__ Q,
                const unsigned short* __restrict__ K,
                const unsigned short* __restrict__ VT,
                unsigned short* __restrict__ O) {
    __shared__ unsigned short lQ[64][LDF];
    __shared__ unsigned short lK[2][64][LDF];
    __shared__ unsigned short lV[2][64][LDF];   // V^T tile: [d][sk]
    __shared__ unsigned short lP[4][16][LDF];   // per-wave P staging

    const int t    = threadIdx.x;
    const int lane = t & 31, w = t >> 5;
    const int half = lane >> 4, lm = lane & 15;
    const int q0   = blockIdx.x * 64;
    const int bh   = blockIdx.y;

    const unsigned short* Kbase = K  + (size_t)bh * S_LEN * DK;
    const unsigned short* Vbase = VT + (size_t)bh * DK * S_LEN;

    // async-prefetch a 64-key tile (K rows + V^T rows) into buffer `buf`
    auto issue_tile = [&](int skt, int buf) {
        const int sk0 = skt * 64;
#pragma unroll
        for (int i = 0; i < 4; ++i) {
            int u = t + i * 128, r = u >> 3, c = u & 7;
            async_cp_b128(&lK[buf][r][c * 8],
                          Kbase + ((size_t)(sk0 + r)) * DK + c * 8);
            async_cp_b128(&lV[buf][r][c * 8],
                          Vbase + (size_t)r * S_LEN + sk0 + c * 8);
        }
    };

    issue_tile(0, 0);

    // load Q tile (regular vector path; overlaps with the async copies)
    const unsigned short* Qb = Q + ((size_t)bh * S_LEN + q0) * DK;
#pragma unroll
    for (int i = 0; i < 4; ++i) {
        int u = t + i * 128, r = u >> 3, c = u & 7;
        *(u32x4*)&lQ[r][c * 8] = *(const u32x4*)&Qb[(size_t)r * DK + c * 8];
    }
    wait_async0();
    __syncthreads();                 // tile 0 + lQ resident for all waves

    v16bf qf[2];
    {
        const unsigned short* row = &lQ[16 * w + lm][0];
        qf[0] = ld_frag(row + half * 8,      row + 16 + half * 8);
        qf[1] = ld_frag(row + 32 + half * 8, row + 48 + half * 8);
    }

    v8f acco[4];
#pragma unroll
    for (int dt = 0; dt < 4; ++dt) acco[dt] = vzero();
    float mrun[8], lrun[8];
#pragma unroll
    for (int j = 0; j < 8; ++j) { mrun[j] = -3.0e38f; lrun[j] = 0.f; }

    const int NT = S_LEN / 64;
    for (int skt = 0; skt < NT; ++skt) {
        const int buf = skt & 1;
        // prefetch next tile into the other buffer (its last readers passed
        // the barrier that ended the previous iteration)
        if (skt + 1 < NT) issue_tile(skt + 1, buf ^ 1);

        // ---- S = Q @ K^T  (lK[sk][d] is already B^T layout) ----
        v8f sc[4];
#pragma unroll
        for (int nt = 0; nt < 4; ++nt) {
            const unsigned short* row = &lK[buf][16 * nt + lm][0];
            v16bf b0 = ld_frag(row + half * 8,      row + 16 + half * 8);
            v16bf b1 = ld_frag(row + 32 + half * 8, row + 48 + half * 8);
            v8f z = vzero();
            z = __builtin_amdgcn_wmma_f32_16x16x32_bf16(false, qf[0], false, b0,
                                                        (short)0, z, false, false);
            sc[nt] = __builtin_amdgcn_wmma_f32_16x16x32_bf16(false, qf[1], false, b1,
                                                             (short)0, z, false, false);
        }
        // scale 1/sqrt(Dk) = 0.125
#pragma unroll
        for (int nt = 0; nt < 4; ++nt)
#pragma unroll
            for (int j = 0; j < 8; ++j) sc[nt][j] *= 0.125f;

        // ---- online softmax (row = j + 8*half, reduce across 16-lane half) ----
        float mnew[8], corr[8];
#pragma unroll
        for (int j = 0; j < 8; ++j) {
            float mx = fmaxf(fmaxf(sc[0][j], sc[1][j]), fmaxf(sc[2][j], sc[3][j]));
#pragma unroll
            for (int off = 1; off < 16; off <<= 1)
                mx = fmaxf(mx, __shfl_xor(mx, off, 32));
            mnew[j] = fmaxf(mrun[j], mx);
            corr[j] = __expf(mrun[j] - mnew[j]);
            mrun[j] = mnew[j];
        }
#pragma unroll
        for (int j = 0; j < 8; ++j) {
            float rs = 0.f;
#pragma unroll
            for (int nt = 0; nt < 4; ++nt) {
                float p = __expf(sc[nt][j] - mnew[j]);
                sc[nt][j] = p;
                rs += p;
            }
#pragma unroll
            for (int off = 1; off < 16; off <<= 1)
                rs += __shfl_xor(rs, off, 32);
            lrun[j] = lrun[j] * corr[j] + rs;
        }
#pragma unroll
        for (int dt = 0; dt < 4; ++dt)
#pragma unroll
            for (int j = 0; j < 8; ++j) acco[dt][j] *= corr[j];

        // ---- P: C-layout -> A-layout via per-wave LDS staging ----
#pragma unroll
        for (int nt = 0; nt < 4; ++nt)
#pragma unroll
            for (int j = 0; j < 8; ++j)
                lP[w][j + 8 * half][16 * nt + lm] = f2bf(sc[nt][j]);
        __syncthreads();
        v16bf pf[2];
        {
            const unsigned short* prow = &lP[w][lm][0];
            pf[0] = ld_frag(prow + half * 8,      prow + 16 + half * 8);
            pf[1] = ld_frag(prow + 32 + half * 8, prow + 48 + half * 8);
        }
        // ---- O += P @ V  (lV[d][sk] is B^T layout) ----
#pragma unroll
        for (int dt = 0; dt < 4; ++dt) {
            const unsigned short* row = &lV[buf][16 * dt + lm][0];
            v16bf b0 = ld_frag(row + half * 8,      row + 16 + half * 8);
            v16bf b1 = ld_frag(row + 32 + half * 8, row + 48 + half * 8);
            acco[dt] = __builtin_amdgcn_wmma_f32_16x16x32_bf16(false, pf[0], false, b0,
                                                               (short)0, acco[dt], false, false);
            acco[dt] = __builtin_amdgcn_wmma_f32_16x16x32_bf16(false, pf[1], false, b1,
                                                               (short)0, acco[dt], false, false);
        }

        // next tile resident + everyone done with this iteration's buffers
        if (skt + 1 < NT) {
            wait_async0();
            __syncthreads();
        }
    }

    // ---- finalize: divide by l, write bf16 to [B*S][1024] with head offset ----
    const int b = bh >> 4, h = bh & 15;
#pragma unroll
    for (int j = 0; j < 8; ++j) {
        float inv = lrun[j] > 0.f ? 1.f / lrun[j] : 0.f;
        int s = q0 + 16 * w + j + 8 * half;
        unsigned short* orow = O + ((size_t)(b * S_LEN + s)) * DM + h * DK;
#pragma unroll
        for (int dt = 0; dt < 4; ++dt)
            orow[16 * dt + lm] = f2bf(acco[dt][j] * inv);
    }
}

// ---------------------------------------------------------------------------
extern "C" void kernel_launch(void* const* d_in, const int* in_sizes, int n_in,
                              void* d_out, int out_size, void* d_ws, size_t ws_size,
                              hipStream_t stream) {
    (void)in_sizes; (void)n_in; (void)out_size; (void)ws_size;
    const float* query  = (const float*)d_in[0];
    const float* key_in = (const float*)d_in[1];
    const float* value  = (const float*)d_in[2];
    const float* w_q = (const float*)d_in[3];
    const float* b_q = (const float*)d_in[4];
    const float* w_k = (const float*)d_in[5];
    const float* b_k = (const float*)d_in[6];
    const float* w_v = (const float*)d_in[7];
    const float* b_v = (const float*)d_in[8];
    const float* w_o = (const float*)d_in[9];
    const float* b_o = (const float*)d_in[10];

    const size_t headElems = (size_t)2 * NH * S_LEN * DK;   // 4 Mi elems
    unsigned short* qws = (unsigned short*)d_ws;
    unsigned short* kws = qws + headElems;
    unsigned short* vws = kws + headElems;
    unsigned short* ows = vws + headElems;

    dim3 g(DM / 128, (2 * S_LEN) / 128);   // (8, 32)
    gemm128<0><<<g, 128, 0, stream>>>(query,  w_q, b_q, qws);
    gemm128<0><<<g, 128, 0, stream>>>(key_in, w_k, b_k, kws);
    gemm128<2><<<g, 128, 0, stream>>>(value,  w_v, b_v, vws);
    flash_attn<<<dim3(S_LEN / 64, 2 * NH), 128, 0, stream>>>(qws, kws, vws, ows);
    gemm128<3><<<g, 128, 0, stream>>>(ows, w_o, b_o, d_out);
}
